// LotteryModel_34943853920856
// MI455X (gfx1250) — matrix operand order repaired and checked by reading, not compiled, not run on hardware
//
#include <hip/hip_runtime.h>
#include <hip/hip_bf16.h>

typedef __bf16 v8bf  __attribute__((ext_vector_type(8)));
typedef __bf16 v16bf __attribute__((ext_vector_type(16)));
typedef float  v8f   __attribute__((ext_vector_type(8)));

#define Nn 512
#define Tt 128
#define Ll 32
#define Dd 17
#define Hh 30
#define NLY 3
#define SS (Tt*Ll)           // 4096 time steps
#define GW 120               // 4*H gate width
#define KPAD 64              // padded K for A staging
#define AELEMS (16*KPAD)     // elements per A staging buffer (1024)
#define GP 128               // gates pitch (fp32 elements)

// ---- workspace layout (bytes); mirrored 1:1 into LDS ----
#define FRAG_OFF    0        // 25600 bf16 B-fragment elements = 51200 B
#define FRAG_ELEMS  25600    // 3 layers*8 ct*2 kt*512 + proj 2*512
#define BSUM_OFF    51200    // 3*128 fp32 fused biases = 1536 B
#define BO_OFF      52736    // 32 fp32 output bias = 128 B
#define WCOPY_BYTES 52864

// ---- LDS-only regions (bytes) ----
#define A_OFF    52864       // 3 x (16x64 bf16 = 2048 B), contiguous
#define P_OFF    59008       // 16x32 bf16 = 1024 B  (projection A staging)
#define G_OFF    60032       // 16x128 f32 = 8192 B  (gates)
#define C_OFF    68224       // 3*16*32 f32 = 6144 B (cell state)
#define LDS_BYTES 74368

// -------------------- setup: swizzle weights into WMMA B-fragment order ----
// B fragment as packed here (contiguous K, matching load_a): lanes 0-15 hold K=0..15, lanes 16-31 hold
// K=16..31; per lane 16 contiguous K values for fixed column N = lane%16.
// B[k][n] = Wcat[n][k] where Wcat = [W_ih ; W_hh] concatenated along K.
__global__ void setup_frags(
    const float* Wih0, const float* Whh0, const float* bih0, const float* bhh0,
    const float* Wih1, const float* Whh1, const float* bih1, const float* bhh1,
    const float* Wih2, const float* Whh2, const float* bih2, const float* bhh2,
    const float* Wo,   const float* bo,   char* ws)
{
    __bf16* frag = (__bf16*)(ws + FRAG_OFF);
    float*  bsum = (float*)(ws + BSUM_OFF);
    float*  bop  = (float*)(ws + BO_OFF);
    int tid  = blockIdx.x * blockDim.x + threadIdx.x;
    int nthr = gridDim.x * blockDim.x;

    for (int e = tid; e < FRAG_ELEMS; e += nthr) {
        float v = 0.0f;
        if (e < 24576) {                       // LSTM layer fragments
            int l    = e >> 13;                // /8192 : layer
            int rem  = e & 8191;
            int ct   = rem >> 10;              // gate-column tile 0..7
            int kt   = (rem >> 9) & 1;         // K tile 0..1
            int li   = rem & 511;
            int lane = li >> 4;
            int elem = li & 15;
            int K = kt * 32 + ((lane >= 16) ? 16 : 0) + elem;
            int n = ct * 16 + (lane & 15);
            int in = (l == 0) ? Dd : Hh;
            const float* Wih = (l == 0) ? Wih0 : ((l == 1) ? Wih1 : Wih2);
            const float* Whh = (l == 0) ? Whh0 : ((l == 1) ? Whh1 : Whh2);
            if (n < GW) {
                if (K < in)            v = Wih[n * in + K];
                else if (K < in + Hh)  v = Whh[n * Hh + (K - in)];
            }
        } else {                               // output projection fragments
            int e2   = e - 24576;
            int ct   = e2 >> 9;                // 0..1
            int li   = e2 & 511;
            int lane = li >> 4;
            int elem = li & 15;
            int K = ((lane >= 16) ? 16 : 0) + elem;
            int n = ct * 16 + (lane & 15);
            if (n < Dd && K < Hh) v = Wo[n * Hh + K];
        }
        frag[e] = (__bf16)v;
    }
    for (int i = tid; i < 3 * GP; i += nthr) {
        int l = i >> 7, col = i & 127;
        const float* bih = (l == 0) ? bih0 : ((l == 1) ? bih1 : bih2);
        const float* bhh = (l == 0) ? bhh0 : ((l == 1) ? bhh1 : bhh2);
        bsum[i] = (col < GW) ? (bih[col] + bhh[col]) : 0.0f;
    }
    for (int i = tid; i < 32; i += nthr) bop[i] = (i < Dd) ? bo[i] : 0.0f;
}

// -------------------- persistent LSTM kernel (one wave = 16 batch rows) ----
__device__ __forceinline__ void lds_fence() {
    asm volatile("s_wait_dscnt 0" ::: "memory");
}
__device__ __forceinline__ float sigm(float v)  { return 1.0f / (1.0f + __expf(-v)); }
__device__ __forceinline__ float tanhx(float v) { float e = __expf(-2.0f * v); return (1.0f - e) / (1.0f + e); }

// A fragment (16-bit, 16x32) from row-major LDS staging:
// lanes 0-15: row=lane, K chunks {kb+0..7, kb+8..15}
// lanes 16-31: row=lane-16, K chunks {kb+16..23, kb+24..31}
template <int PITCH>
__device__ __forceinline__ v16bf load_a(const __bf16* base, int kb, int lane) {
    int row = lane & 15;
    const __bf16* p = base + row * PITCH + kb + ((lane & 16) ? 16 : 0);
    v8bf lo = *(const v8bf*)p;
    v8bf hi = *(const v8bf*)(p + 8);
    return __builtin_shufflevector(lo, hi, 0,1,2,3,4,5,6,7,8,9,10,11,12,13,14,15);
}
__device__ __forceinline__ v16bf load_b(const __bf16* frag, int fi, int lane) {
    return *(const v16bf*)(frag + fi * 512 + lane * 16);
}

__global__ void __launch_bounds__(32)
lstm_persistent(const float* __restrict__ x, const float* __restrict__ h0,
                const float* __restrict__ c0, const char* __restrict__ ws,
                float* __restrict__ out, float* __restrict__ hout,
                float* __restrict__ cout)
{
    extern __shared__ char smem[];
    const int lane  = threadIdx.x;
    const int nbase = blockIdx.x * 16;

    __bf16* FRAG  = (__bf16*)(smem + FRAG_OFF);
    float*  BSUM  = (float*)(smem + BSUM_OFF);
    float*  BO    = (float*)(smem + BO_OFF);
    __bf16* ABASE = (__bf16*)(smem + A_OFF);   // layer l buffer = ABASE + l*AELEMS
    __bf16* P     = (__bf16*)(smem + P_OFF);
    float*  G     = (float*)(smem + G_OFF);
    float*  C     = (float*)(smem + C_OFF);

    // 1) copy swizzled weights + biases into LDS (once)
    {
        const uint32_t* src = (const uint32_t*)ws;
        uint32_t*       dst = (uint32_t*)smem;
        for (int i = lane; i < WCOPY_BYTES / 4; i += 32) dst[i] = src[i];
    }
    // 2) zero all A staging + projection staging (K padding stays zero)
    {
        uint32_t* z = (uint32_t*)(smem + A_OFF);
        for (int i = lane; i < (P_OFF + 1024 - A_OFF) / 4; i += 32) z[i] = 0u;
    }
    // 3) seed recurrent h slots and cell state from h0/c0
    const int hoff0 = Dd;   // layer0 recurrent slot starts after x
    for (int idx = lane; idx < NLY * 16 * Hh; idx += 32) {
        int l = idx / 480, rem = idx % 480;
        int row = rem / Hh, u = rem % Hh;
        size_t gi = ((size_t)l * Nn + nbase + row) * Hh + u;
        int off = (l == 0) ? hoff0 : Hh;
        ABASE[l * AELEMS + row * KPAD + off + u] = (__bf16)h0[gi];
        C[(l * 16 + row) * 32 + u] = c0[gi];
    }
    lds_fence();

    const int ccol = lane & 15;
    const int crow = (lane & 16) ? 8 : 0;

    for (int s = 0; s < SS; ++s) {
        // stage x_t -> layer0 A input slot (cols 0..16)
        for (int idx = lane; idx < 16 * Dd; idx += 32) {
            int row = idx / Dd, col = idx % Dd;
            ABASE[row * KPAD + col] =
                (__bf16)x[(((size_t)(nbase + row)) * SS + s) * Dd + col];
        }
        lds_fence();

        for (int l = 0; l < NLY; ++l) {
            __bf16* Al = ABASE + l * AELEMS;
            v16bf a0 = load_a<KPAD>(Al, 0, lane);
            v16bf a1 = load_a<KPAD>(Al, 32, lane);
            #pragma unroll
            for (int ct = 0; ct < 8; ++ct) {
                float bb = BSUM[l * GP + ct * 16 + ccol];
                v8f acc = { bb, bb, bb, bb, bb, bb, bb, bb };
                v16bf b0 = load_b(FRAG, (l * 8 + ct) * 2 + 0, lane);
                v16bf b1 = load_b(FRAG, (l * 8 + ct) * 2 + 1, lane);
                acc = __builtin_amdgcn_wmma_f32_16x16x32_bf16(
                          false, a0, false, b0, (short)0, acc, false, false);
                acc = __builtin_amdgcn_wmma_f32_16x16x32_bf16(
                          false, a1, false, b1, (short)0, acc, false, false);
                #pragma unroll
                for (int r = 0; r < 8; ++r)
                    G[(crow + r) * GP + ct * 16 + ccol] = acc[r];
            }
            lds_fence();

            // elementwise gate update: 16 rows x 30 units = 480 = 15 passes
            int off = (l == 0) ? hoff0 : Hh;
            for (int idx = lane; idx < 16 * Hh; idx += 32) {
                int row = idx / Hh, u = idx % Hh;
                float gi = G[row * GP + u];
                float gf = G[row * GP + Hh + u];
                float gg = G[row * GP + 2 * Hh + u];
                float go = G[row * GP + 3 * Hh + u];
                float cold = C[(l * 16 + row) * 32 + u];
                float cn = sigm(gf) * cold + sigm(gi) * tanhx(gg);
                float hn = sigm(go) * tanhx(cn);
                C[(l * 16 + row) * 32 + u] = cn;
                __bf16 hb = (__bf16)hn;
                Al[row * KPAD + off + u] = hb;                      // recurrent (t+1)
                if (l < 2) ABASE[(l + 1) * AELEMS + row * KPAD + u] = hb; // next layer in
                else       P[row * 32 + u] = hb;                          // projection in
            }
            lds_fence();
        }

        // output projection: (16x30)x(30x17), K padded to 32
        {
            v16bf pa = load_a<32>(P, 0, lane);
            #pragma unroll
            for (int ct = 0; ct < 2; ++ct) {
                float bb = BO[ct * 16 + ccol];
                v8f acc = { bb, bb, bb, bb, bb, bb, bb, bb };
                v16bf bf = load_b(FRAG, 48 + ct, lane);
                acc = __builtin_amdgcn_wmma_f32_16x16x32_bf16(
                          false, pa, false, bf, (short)0, acc, false, false);
                int col = ct * 16 + ccol;
                if (col < Dd) {
                    #pragma unroll
                    for (int r = 0; r < 8; ++r)
                        out[(((size_t)(nbase + crow + r)) * SS + s) * Dd + col] = acc[r];
                }
            }
        }
    }

    // final h, c outputs
    for (int idx = lane; idx < NLY * 16 * Hh; idx += 32) {
        int l = idx / 480, rem = idx % 480;
        int row = rem / Hh, u = rem % Hh;
        size_t gi = ((size_t)l * Nn + nbase + row) * Hh + u;
        int off = (l == 0) ? hoff0 : Hh;
        hout[gi] = (float)ABASE[l * AELEMS + row * KPAD + off + u];
        cout[gi] = C[(l * 16 + row) * 32 + u];
    }
}

// -------------------------------------------------------------------------
extern "C" void kernel_launch(void* const* d_in, const int* in_sizes, int n_in,
                              void* d_out, int out_size, void* d_ws, size_t ws_size,
                              hipStream_t stream) {
    (void)in_sizes; (void)n_in; (void)out_size; (void)ws_size;
    const float* x    = (const float*)d_in[0];
    const float* h0   = (const float*)d_in[1];
    const float* c0   = (const float*)d_in[2];
    const float* Wih0 = (const float*)d_in[3];
    const float* Whh0 = (const float*)d_in[4];
    const float* bih0 = (const float*)d_in[5];
    const float* bhh0 = (const float*)d_in[6];
    const float* Wih1 = (const float*)d_in[7];
    const float* Whh1 = (const float*)d_in[8];
    const float* bih1 = (const float*)d_in[9];
    const float* bhh1 = (const float*)d_in[10];
    const float* Wih2 = (const float*)d_in[11];
    const float* Whh2 = (const float*)d_in[12];
    const float* bih2 = (const float*)d_in[13];
    const float* bhh2 = (const float*)d_in[14];
    const float* Wo   = (const float*)d_in[15];
    const float* bo   = (const float*)d_in[16];

    float* out  = (float*)d_out;
    float* hout = out + (size_t)Nn * SS * Dd;
    float* cout = hout + (size_t)NLY * Nn * Hh;

    setup_frags<<<dim3(64), dim3(256), 0, stream>>>(
        Wih0, Whh0, bih0, bhh0, Wih1, Whh1, bih1, bhh1,
        Wih2, Whh2, bih2, bhh2, Wo, bo, (char*)d_ws);

    lstm_persistent<<<dim3(Nn / 16), dim3(32), LDS_BYTES, stream>>>(
        x, h0, c0, (const char*)d_ws, out, hout, cout);
}
